// MultiAttentionGATLayer_56049323213778
// MI455X (gfx1250) — matrix operand order repaired and checked by reading, not compiled, not run on hardware
//
#include <hip/hip_runtime.h>

// ---------------------------------------------------------------------------
// MultiAttentionGATLayer for MI455X (gfx1250, wave32, WMMA)
// All large matmuls go through a 128x128x32-tiled bf16 WMMA GEMM
// (fp32 -> bf16 pack via v_perm_b32, fp32 accumulation, reg double-buffer).
// ---------------------------------------------------------------------------

typedef __attribute__((ext_vector_type(16))) __bf16 v16bf;
typedef __attribute__((ext_vector_type(8)))  float  v8f;

#define TILE_M 128
#define TILE_N 128
#define TILE_K 32
#define LDS_STRIDE 40   // ushorts per LDS row (padded: 80B, 16B-aligned chunks)

// pack two f32 into bf16x2 (round-half-up) with a single v_perm_b32
__device__ __forceinline__ unsigned int pack_bf16(float x, float y) {
  union { float f; unsigned int u; } a, b;
  a.f = x; b.f = y;
  // result bytes: [y.b3, y.b2, x.b3, x.b2]  (idx 4-7 = first arg, 0-3 = second)
  return __builtin_amdgcn_perm(b.u + 0x8000u, a.u + 0x8000u, 0x07060302u);
}
__device__ __forceinline__ unsigned short f2bf(float f) {
  union { float f; unsigned int u; } v; v.f = f;
  return (unsigned short)((v.u + 0x8000u) >> 16);
}

// A fragment (16x32 bf16): lane-half h: elems 0..7 -> K = 8h..8h+7,
// elems 8..15 -> K = 16+8h..23+8h   (ISA 7.12.2, 16-bit A matrix 16x32)
__device__ __forceinline__ v16bf load_frag_a(const unsigned short* rowp, int h) {
  union { uint4 u[2]; v16bf v; } t;
  t.u[0] = *(const uint4*)(rowp + 8 * h);
  t.u[1] = *(const uint4*)(rowp + 16 + 8 * h);
  return t.v;
}

// B fragment (32x16 bf16): lane-half h: elems 0..7 -> K = 8h..8h+7, elems 8..15 -> K = 16+8h..23+8h
__device__ __forceinline__ v16bf load_frag_b(const unsigned short* rowp, int h) {
  union { uint4 u[2]; v16bf v; } t;
  t.u[0] = *(const uint4*)(rowp + 8 * h);
  t.u[1] = *(const uint4*)(rowp + 16 + 8 * h);
  return t.v;
}

// C = A(MxK) * B(KxN or NxK if transB) [+ bias], batched over grid.z.
// M,N multiples of 128; K multiple of 32. Row-major fp32 in/out.
__global__ __launch_bounds__(256, 1)
void gemm_wmma_bf16(const float* __restrict__ A, const float* __restrict__ Bm,
                    float* __restrict__ C, const float* __restrict__ bias,
                    int K, long long lda, long long ldb, long long ldc,
                    long long sA, long long sB, long long sC, int transB) {
  __shared__ unsigned short As[TILE_M * LDS_STRIDE];
  __shared__ unsigned short Bs[TILE_N * LDS_STRIDE];

  const int bz = blockIdx.z;
  A  += (long long)bz * sA;
  Bm += (long long)bz * sB;
  C  += (long long)bz * sC;

  const int row0 = blockIdx.x * TILE_M;
  const int col0 = blockIdx.y * TILE_N;
  const int tid  = threadIdx.x;
  const int lane = tid & 31;
  const int wave = tid >> 5;
  const int wm = (wave & 1) * 64;    // wave M offset within block tile
  const int wn = (wave >> 1) * 32;   // wave N offset within block tile
  const int lh = lane >> 4;          // lane half
  const int ln = lane & 15;

  // staging addresses (row-copy pattern used by A always, and B when transB)
  const int rr  = tid >> 1;
  const int rkk = (tid & 1) * 16;
  const float* aSrc = A + (long long)(row0 + rr) * lda + rkk;
  const float* bSrcT = Bm + (long long)(col0 + rr) * ldb + rkk;   // transB path

  const v8f vzero = {0.f, 0.f, 0.f, 0.f, 0.f, 0.f, 0.f, 0.f};
  v8f acc[4][2];
#pragma unroll
  for (int i = 0; i < 4; ++i)
#pragma unroll
    for (int j = 0; j < 2; ++j) acc[i][j] = vzero;

  float4 pfA[4], pfB[4];

  // ---- prologue: prefetch tile k0 = 0 into registers ----
#pragma unroll
  for (int j = 0; j < 4; ++j) pfA[j] = *(const float4*)(aSrc + 4 * j);
  if (transB) {
#pragma unroll
    for (int j = 0; j < 4; ++j) pfB[j] = *(const float4*)(bSrcT + 4 * j);
  } else {
#pragma unroll
    for (int i = 0; i < 4; ++i) {
      const int l  = tid + 256 * i;
      const int kk = l >> 5;
      const int nq = (l & 31) * 4;
      pfB[i] = *(const float4*)(Bm + (long long)kk * ldb + col0 + nq);
    }
  }

  for (int k0 = 0; k0 < K; k0 += TILE_K) {
    // ---- stage prefetched registers -> LDS (bf16) ----
    {
      uint4 u0, u1;
      u0.x = pack_bf16(pfA[0].x, pfA[0].y); u0.y = pack_bf16(pfA[0].z, pfA[0].w);
      u0.z = pack_bf16(pfA[1].x, pfA[1].y); u0.w = pack_bf16(pfA[1].z, pfA[1].w);
      u1.x = pack_bf16(pfA[2].x, pfA[2].y); u1.y = pack_bf16(pfA[2].z, pfA[2].w);
      u1.z = pack_bf16(pfA[3].x, pfA[3].y); u1.w = pack_bf16(pfA[3].z, pfA[3].w);
      uint4* dst = (uint4*)&As[rr * LDS_STRIDE + rkk];
      dst[0] = u0; dst[1] = u1;
    }
    if (transB) {
      uint4 u0, u1;
      u0.x = pack_bf16(pfB[0].x, pfB[0].y); u0.y = pack_bf16(pfB[0].z, pfB[0].w);
      u0.z = pack_bf16(pfB[1].x, pfB[1].y); u0.w = pack_bf16(pfB[1].z, pfB[1].w);
      u1.x = pack_bf16(pfB[2].x, pfB[2].y); u1.y = pack_bf16(pfB[2].z, pfB[2].w);
      u1.z = pack_bf16(pfB[3].x, pfB[3].y); u1.w = pack_bf16(pfB[3].z, pfB[3].w);
      uint4* dst = (uint4*)&Bs[rr * LDS_STRIDE + rkk];
      dst[0] = u0; dst[1] = u1;
    } else {
#pragma unroll
      for (int i = 0; i < 4; ++i) {
        const int l  = tid + 256 * i;
        const int kk = l >> 5;
        const int nq = (l & 31) * 4;
        Bs[(nq + 0) * LDS_STRIDE + kk] = f2bf(pfB[i].x);
        Bs[(nq + 1) * LDS_STRIDE + kk] = f2bf(pfB[i].y);
        Bs[(nq + 2) * LDS_STRIDE + kk] = f2bf(pfB[i].z);
        Bs[(nq + 3) * LDS_STRIDE + kk] = f2bf(pfB[i].w);
      }
    }
    __syncthreads();

    // ---- issue next tile's global loads (overlap with WMMA below) ----
    const int kn = k0 + TILE_K;
    if (kn < K) {
#pragma unroll
      for (int j = 0; j < 4; ++j) pfA[j] = *(const float4*)(aSrc + kn + 4 * j);
      if (transB) {
#pragma unroll
        for (int j = 0; j < 4; ++j) pfB[j] = *(const float4*)(bSrcT + kn + 4 * j);
      } else {
#pragma unroll
        for (int i = 0; i < 4; ++i) {
          const int l  = tid + 256 * i;
          const int kk = l >> 5;
          const int nq = (l & 31) * 4;
          pfB[i] = *(const float4*)(Bm + (long long)(kn + kk) * ldb + col0 + nq);
        }
      }
    }

    // ---- WMMA compute on staged tile ----
    v16bf af[4], bfr[2];
#pragma unroll
    for (int mt = 0; mt < 4; ++mt)
      af[mt] = load_frag_a(&As[(wm + mt * 16 + ln) * LDS_STRIDE], lh);
#pragma unroll
    for (int nt = 0; nt < 2; ++nt)
      bfr[nt] = load_frag_b(&Bs[(wn + nt * 16 + ln) * LDS_STRIDE], lh);

#pragma unroll
    for (int mt = 0; mt < 4; ++mt)
#pragma unroll
      for (int nt = 0; nt < 2; ++nt)
        acc[mt][nt] = __builtin_amdgcn_wmma_f32_16x16x32_bf16(
            false, af[mt], false, bfr[nt], (short)0, acc[mt][nt], false, false);

    __syncthreads();
  }

  // ---- epilogue: D layout (ISA 7.12.2): VGPR v -> M = v + 8*laneHalf, N = lane%16
#pragma unroll
  for (int mt = 0; mt < 4; ++mt)
#pragma unroll
    for (int nt = 0; nt < 2; ++nt) {
      const int col = col0 + wn + nt * 16 + ln;
      const float bv = bias ? bias[col] : 0.f;
#pragma unroll
      for (int v = 0; v < 8; ++v) {
        const int row = row0 + wm + mt * 16 + v + 8 * lh;
        C[(long long)row * ldc + col] = acc[mt][nt][v] + bv;
      }
    }
}

// ---------------------------------------------------------------------------
// helpers: block reductions (256 threads)
// ---------------------------------------------------------------------------
__device__ __forceinline__ float blk_sum(float v, float* sm) {
  const int t = threadIdx.x;
  sm[t] = v; __syncthreads();
  for (int s = 128; s > 0; s >>= 1) { if (t < s) sm[t] += sm[t + s]; __syncthreads(); }
  float r = sm[0]; __syncthreads(); return r;
}
__device__ __forceinline__ float blk_max(float v, float* sm) {
  const int t = threadIdx.x;
  sm[t] = v; __syncthreads();
  for (int s = 128; s > 0; s >>= 1) { if (t < s) sm[t] = fmaxf(sm[t], sm[t + s]); __syncthreads(); }
  float r = sm[0]; __syncthreads(); return r;
}

// cbuf = { meta_w0, meta_w1, sum(a2), bias_gat[0] }
__global__ void prep_kernel(const float* __restrict__ meta, const float* __restrict__ a2,
                            const float* __restrict__ bg, float* __restrict__ cbuf, int n_a2) {
  __shared__ float sm[256];
  const int t = threadIdx.x;
  float s = 0.f;
  for (int j = t; j < n_a2; j += 256) s += a2[j];
  s = blk_sum(s, sm);
  if (t == 0) {
    float m0 = meta[0], m1 = meta[1];
    float mx = fmaxf(m0, m1);
    float e0 = __expf(m0 - mx), e1 = __expf(m1 - mx);
    float inv = 1.f / (e0 + e1);
    cbuf[0] = e0 * inv; cbuf[1] = e1 * inv; cbuf[2] = s; cbuf[3] = bg[0];
  }
}

// s_src[b,i] = Wh[b,i,:] . a1[0:D] ; s_dst[b,i] = Wh[b,i,:] . a1[D:2D]
__global__ void rowdot_kernel(const float* __restrict__ Wh, const float* __restrict__ a1,
                              float* __restrict__ svec, long long bnTotal, int nNodes, int dim) {
  __shared__ float sm[512];
  const int b = blockIdx.y, i = blockIdx.x, t = threadIdx.x;
  const float* row = Wh + ((long long)b * nNodes + i) * (long long)dim;
  float s1 = 0.f, s2 = 0.f;
  for (int j = t; j < dim; j += 256) { float w = row[j]; s1 += w * a1[j]; s2 += w * a1[dim + j]; }
  sm[t] = s1; sm[256 + t] = s2; __syncthreads();
  for (int s = 128; s > 0; s >>= 1) {
    if (t < s) { sm[t] += sm[t + s]; sm[256 + t] += sm[256 + t + s]; }
    __syncthreads();
  }
  if (t == 0) {
    svec[(long long)b * nNodes + i]           = sm[0];
    svec[bnTotal + (long long)b * nNodes + i] = sm[256];
  }
}

// Fused GAT score combine + adjacency mask + row softmax. Assumes nNodes == 2048.
__global__ void gat_attn_kernel(const float* __restrict__ e2, const float* __restrict__ Wh,
                                const float* __restrict__ svec, const int* __restrict__ adj,
                                const float* __restrict__ cbuf, float* __restrict__ attn,
                                long long bnTotal, int nNodes, int dim) {
  __shared__ float sm[256];
  const int b = blockIdx.y, i = blockIdx.x, t = threadIdx.x;
  const float w0 = cbuf[0], w1 = cbuf[1], a2s = cbuf[2], bgv = cbuf[3];
  const long long rowoff = ((long long)b * nNodes + i);
  const float* e2row  = e2 + rowoff * (long long)nNodes;
  const float* whrow  = Wh + rowoff * (long long)dim;   // dim == nNodes (e2_bias replication)
  const int*   adjrow = adj + rowoff * (long long)nNodes;
  const float* sdst   = svec + bnTotal + (long long)b * nNodes;
  const float ssrc = svec[(long long)b * nNodes + i];

  float vals[8];
  float lmax = -3.4e38f;
#pragma unroll
  for (int u = 0; u < 8; ++u) {
    const int j = t + u * 256;
    float e1 = ssrc + sdst[j];
    e1 = (e1 >= 0.f) ? e1 : 0.2f * e1;               // LeakyReLU
    float e2v = e2row[j] + whrow[j] * a2s + bgv;
    float e = w0 * e1 + w1 * e2v;
    e = (adjrow[j] == 0) ? -9e15f : e;
    vals[u] = e;
    lmax = fmaxf(lmax, e);
  }
  const float rmax = blk_max(lmax, sm);
  float lsum = 0.f;
#pragma unroll
  for (int u = 0; u < 8; ++u) { vals[u] = __expf(vals[u] - rmax); lsum += vals[u]; }
  const float inv = 1.f / blk_sum(lsum, sm);
  float* arow = attn + rowoff * (long long)nNodes;
#pragma unroll
  for (int u = 0; u < 8; ++u) arow[t + u * 256] = vals[u] * inv;
}

// In-place scaled row softmax over nNodes==2048 columns (MHA scores).
__global__ void mha_softmax_kernel(float* __restrict__ sc, int nNodes, float scale) {
  __shared__ float sm[256];
  const int b = blockIdx.y, i = blockIdx.x, t = threadIdx.x;
  float* row = sc + ((long long)b * nNodes + i) * (long long)nNodes;
  float vals[8];
  float lmax = -3.4e38f;
#pragma unroll
  for (int u = 0; u < 8; ++u) {
    vals[u] = row[t + u * 256] * scale;
    lmax = fmaxf(lmax, vals[u]);
  }
  const float rmax = blk_max(lmax, sm);
  float lsum = 0.f;
#pragma unroll
  for (int u = 0; u < 8; ++u) { vals[u] = __expf(vals[u] - rmax); lsum += vals[u]; }
  const float inv = 1.f / blk_sum(lsum, sm);
#pragma unroll
  for (int u = 0; u < 8; ++u) row[t + u * 256] = vals[u] * inv;
}

// out = x + 0.5*LN(h_prime) + 0.5*LN(mha_out)
__global__ void final_fuse_kernel(const float* __restrict__ x, const float* __restrict__ hp,
                                  const float* __restrict__ mo,
                                  const float* __restrict__ g1, const float* __restrict__ b1,
                                  const float* __restrict__ g2, const float* __restrict__ b2,
                                  float* __restrict__ out, int nNodes, int dim) {
  __shared__ float sm[256];
  const int b = blockIdx.y, i = blockIdx.x, t = threadIdx.x;
  const long long off = ((long long)b * nNodes + i) * (long long)dim;
  const float* h = hp + off;
  const float* m = mo + off;
  const float* xr = x + off;
  float s1 = 0.f, q1 = 0.f, s2 = 0.f, q2 = 0.f;
  for (int j = t; j < dim; j += 256) {
    float a = h[j]; s1 += a; q1 += a * a;
    float c = m[j]; s2 += c; q2 += c * c;
  }
  s1 = blk_sum(s1, sm); q1 = blk_sum(q1, sm);
  s2 = blk_sum(s2, sm); q2 = blk_sum(q2, sm);
  const float invD = 1.f / (float)dim;
  const float mu1 = s1 * invD, var1 = q1 * invD - mu1 * mu1;
  const float mu2 = s2 * invD, var2 = q2 * invD - mu2 * mu2;
  const float rs1 = rsqrtf(var1 + 1e-5f);
  const float rs2 = rsqrtf(var2 + 1e-5f);
  float* o = out + off;
  for (int j = t; j < dim; j += 256) {
    float ln1 = (h[j] - mu1) * rs1 * g1[j] + b1[j];
    float ln2 = (m[j] - mu2) * rs2 * g2[j] + b2[j];
    o[j] = xr[j] + 0.5f * ln1 + 0.5f * ln2;
  }
}

// ---------------------------------------------------------------------------
extern "C" void kernel_launch(void* const* d_in, const int* in_sizes, int n_in,
                              void* d_out, int out_size, void* d_ws, size_t ws_size,
                              hipStream_t stream) {
  (void)in_sizes; (void)n_in; (void)out_size; (void)ws_size;

  const float* x    = (const float*)d_in[0];
  const int*   adj  = (const int*)d_in[1];
  const float* W    = (const float*)d_in[2];
  const float* a1   = (const float*)d_in[3];
  const float* U    = (const float*)d_in[4];
  const float* a2   = (const float*)d_in[5];
  const float* bg   = (const float*)d_in[6];
  const float* meta = (const float*)d_in[7];
  const float* q_w  = (const float*)d_in[8];
  const float* q_b  = (const float*)d_in[9];
  const float* k_w  = (const float*)d_in[10];
  const float* k_b  = (const float*)d_in[11];
  const float* v_w  = (const float*)d_in[12];
  const float* v_b  = (const float*)d_in[13];
  const float* o_w  = (const float*)d_in[14];
  const float* o_b  = (const float*)d_in[15];
  const float* ln1g = (const float*)d_in[16];
  const float* ln1b = (const float*)d_in[17];
  const float* ln2g = (const float*)d_in[18];
  const float* ln2b = (const float*)d_in[19];
  float* out = (float*)d_out;

  const int Bb = 4, N = 2048, D = 2048, HEADS = 4, HDm = 512;
  const long long NBb = (long long)N * D;      // per-batch plane (4M floats)
  const long long NNb = (long long)N * N;
  const long long bnTot = (long long)Bb * N;

  float* ws0  = (float*)d_ws;
  float* cbuf = ws0;                 // 4 floats
  float* svec = ws0 + 64;            // 2*B*N floats
  float* big  = ws0 + 32768;
  float* Wh = big + 0 * (long long)Bb * NBb;   // Wh, later reused as q
  float* Tb = big + 1 * (long long)Bb * NBb;   // e2_1, later attn, later k
  float* E2 = big + 2 * (long long)Bb * NBb;   // e2, later v
  float* HP = big + 3 * (long long)Bb * NBb;   // h_prime
  float* SC = big + 4 * (long long)Bb * NBb;   // scores, later mha_out
  float* AO = big + 5 * (long long)Bb * NBb;   // attention output (pre o_w)

  dim3 blk(256);
  auto gemm = [&](const float* A, const float* Bm, float* C, const float* bias,
                  int M, int Ncols, int K, long long lda, long long ldb, long long ldc,
                  long long sA, long long sB, long long sC, int transB) {
    dim3 grid(M / TILE_M, Ncols / TILE_N, Bb);
    gemm_wmma_bf16<<<grid, blk, 0, stream>>>(A, Bm, C, bias, K, lda, ldb, ldc,
                                             sA, sB, sC, transB);
  };

  // consts: meta softmax, sum(a2), bias_gat
  prep_kernel<<<1, blk, 0, stream>>>(meta, a2, bg, cbuf, 2 * D);

  // Wh = x @ W                       (B x) 2048x2048x2048
  gemm(x, W, Wh, nullptr, N, D, D, D, D, D, NBb, 0, NBb, 0);

  // s_src / s_dst row dots
  rowdot_kernel<<<dim3(N, Bb), blk, 0, stream>>>(Wh, a1, svec, bnTot, N, D);

  // T = Wh @ U
  gemm(Wh, U, Tb, nullptr, N, D, D, D, D, D, NBb, 0, NBb, 0);

  // e2 = T @ Wh^T
  gemm(Tb, Wh, E2, nullptr, N, N, D, D, D, N, NBb, NBb, NNb, 1);

  // attention = softmax(mask(w0*leaky(e1) + w1*(e2 + bias terms)))  -> Tb
  gat_attn_kernel<<<dim3(N, Bb), blk, 0, stream>>>(E2, Wh, svec, adj, cbuf, Tb, bnTot, N, D);

  // h_prime = attention @ Wh
  gemm(Tb, Wh, HP, nullptr, N, D, N, N, D, D, NNb, NBb, NBb, 0);

  // q/k/v projections (overwrite Wh / Tb / E2)
  gemm(HP, q_w, Wh, q_b, N, D, D, D, D, D, NBb, 0, NBb, 0);
  gemm(HP, k_w, Tb, k_b, N, D, D, D, D, D, NBb, 0, NBb, 0);
  gemm(HP, v_w, E2, v_b, N, D, D, D, D, D, NBb, 0, NBb, 0);

  // inner MHA: per head, scores = q@k^T * hd^-0.5 ; softmax ; ao = aw@v
  const float scale = 0.044194173824159216f;   // 512^-0.5
  for (int h = 0; h < HEADS; ++h) {
    gemm(Wh + h * HDm, Tb + h * HDm, SC, nullptr, N, N, HDm, D, D, N, NBb, NBb, NNb, 1);
    mha_softmax_kernel<<<dim3(N, Bb), blk, 0, stream>>>(SC, N, scale);
    gemm(SC, E2 + h * HDm, AO + h * HDm, nullptr, N, HDm, N, N, D, D, NNb, NBb, NBb, 0);
  }

  // mha_out = ao @ o_w + o_b   (into SC)
  gemm(AO, o_w, SC, o_b, N, D, D, D, D, D, NBb, 0, NBb, 0);

  // out = x + 0.5*LN1(h_prime) + 0.5*LN2(mha_out)
  final_fuse_kernel<<<dim3(N, Bb), blk, 0, stream>>>(x, HP, SC, ln1g, ln1b, ln2g, ln2b,
                                                     out, N, D);
}
